// AttnBlock_4982162063445
// MI455X (gfx1250) — matrix operand order repaired and checked by reading, not compiled, not run on hardware
//
#include <hip/hip_runtime.h>
#include <hip/hip_bf16.h>
#include <math.h>

// ---------------------------------------------------------------------------
// AttnBlock for B=8, C=256, H=W=64 (N=4096) on gfx1250 (wave32, WMMA + TDM).
// gn_stats -> wcvt -> hnorm(f16 [B,N,C]) -> qk gemm -> vT gemm
//   -> flash-attn (WMMA f16, double-buffered TDM staging) -> proj+residual
// ---------------------------------------------------------------------------

#define BATCH 8
#define CCH   256
#define NTOK  4096
#define NGRP  32

typedef _Float16 h16;
typedef _Float16 v8h  __attribute__((ext_vector_type(8)));
typedef _Float16 v16h __attribute__((ext_vector_type(16)));
typedef float    v8f  __attribute__((ext_vector_type(8)));
typedef unsigned int v4u __attribute__((ext_vector_type(4)));
typedef int      v4i  __attribute__((ext_vector_type(4)));
typedef int      v8i  __attribute__((ext_vector_type(8)));

__device__ __forceinline__ v16h cat16(v8h a, v8h b) {
  return __builtin_shufflevector(a, b, 0,1,2,3,4,5,6,7,8,9,10,11,12,13,14,15);
}

// A fragment (16x32 f16, M x K): lanes 0-15 row L K{0..7,16..23},
// lanes 16-31 row L-16 K{8..15,24..31}.
__device__ __forceinline__ v16h load_a_frag(const h16* __restrict__ src, int ld, int lane) {
  const int r  = lane & 15;
  const int hi = lane >> 4;
  const h16* p = src + r * ld + hi * 8;
  v8h lo = *(const v8h*)(p);
  v8h hh = *(const v8h*)(p + 16);
  return cat16(lo, hh);
}

// B fragment (32x16 f16, K x N): column n at src + n*ld; lanes 0-15 K{0..7,16..23},
// lanes 16-31 K{8..15,24..31}.
__device__ __forceinline__ v16h load_b_frag(const h16* __restrict__ src, int ld, int lane) {
  const int c  = lane & 15;
  const int hi = lane >> 4;
  const h16* p = src + c * ld + hi * 8;
  v8h lo = *(const v8h*)(p);
  v8h hh = *(const v8h*)(p + 16);
  return cat16(lo, hh);
}

__device__ __forceinline__ v8f wmma_f16(v16h a, v16h b, v8f c) {
  return __builtin_amdgcn_wmma_f32_16x16x32_f16(false, a, false, b, (short)0, c, false, false);
}

__device__ __forceinline__ float half_max16(float v) {
  v = fmaxf(v, __shfl_xor(v, 1, 32));
  v = fmaxf(v, __shfl_xor(v, 2, 32));
  v = fmaxf(v, __shfl_xor(v, 4, 32));
  v = fmaxf(v, __shfl_xor(v, 8, 32));
  return v;
}
__device__ __forceinline__ float half_sum16(float v) {
  v += __shfl_xor(v, 1, 32);
  v += __shfl_xor(v, 2, 32);
  v += __shfl_xor(v, 4, 32);
  v += __shfl_xor(v, 8, 32);
  return v;
}

// ---------------------------------------------------------------------------
// Tensor Data Mover: 2D tile load global -> LDS with LDS row padding.
// ---------------------------------------------------------------------------
#if defined(__HIP_DEVICE_COMPILE__) && __has_builtin(__builtin_amdgcn_tensor_load_to_lds)
#define USE_TDM 1
#endif

#ifdef USE_TDM
__device__ __forceinline__ unsigned lds_addr_of(const void* p) {
  return (unsigned)(unsigned long long)(__attribute__((address_space(3))) const void*)p;
}

__device__ __forceinline__ void tdm_load_2d(unsigned lds_byte, const void* gaddr,
                                            unsigned tile0, unsigned tile1,
                                            unsigned tdim0, unsigned tdim1,
                                            unsigned stride0,
                                            unsigned pad_interval, unsigned pad_amount) {
  unsigned long long ga = (unsigned long long)gaddr;
  v4u g0;
  g0.x = 1u;                                                 // count=1, user mode
  g0.y = lds_byte;                                           // lds_addr
  g0.z = (unsigned)(ga & 0xFFFFFFFFu);                       // global_addr[31:0]
  g0.w = (unsigned)((ga >> 32) & 0x01FFFFFFu) | 0x80000000u; // [56:32] | type=2
  v8i g1;
  unsigned d0 = (1u << 16);                                  // data_size = 2 bytes
  if (pad_amount) d0 |= (1u << 20) | (pad_interval << 22) | (pad_amount << 25);
  g1[0] = (int)d0;
  g1[1] = (int)((tdim0 & 0xFFFFu) << 16);
  g1[2] = (int)((tdim0 >> 16) | ((tdim1 & 0xFFFFu) << 16));
  g1[3] = (int)((tdim1 >> 16) | (tile0 << 16));
  g1[4] = (int)(tile1 & 0xFFFFu);
  g1[5] = (int)stride0;
  g1[6] = 0;
  g1[7] = 0;
  v4i z4 = (v4i)0;
#if __clang_major__ >= 23
  v8i z8 = (v8i)0;
  __builtin_amdgcn_tensor_load_to_lds(g0, g1, z4, z4, z8, 0);
#else
  __builtin_amdgcn_tensor_load_to_lds(g0, g1, z4, z4, 0);
#endif
}
#endif

// ---------------------------------------------------------------------------
// K1: group-norm stats.
// ---------------------------------------------------------------------------
__global__ void k_gn_stats(const float* __restrict__ x,
                           float* __restrict__ meanv, float* __restrict__ rstdv) {
  const int b = blockIdx.x / NGRP;
  const int g = blockIdx.x % NGRP;
  const float* p = x + ((size_t)b * CCH + g * 8) * NTOK;
  const int tot = 8 * NTOK;
  float s = 0.f, ss = 0.f;
  for (int i = threadIdx.x; i < tot; i += 256) {
    float v = p[i];
    s += v; ss += v * v;
  }
  __shared__ float sh1[256], sh2[256];
  sh1[threadIdx.x] = s; sh2[threadIdx.x] = ss;
  __syncthreads();
  for (int d = 128; d > 0; d >>= 1) {
    if (threadIdx.x < d) { sh1[threadIdx.x] += sh1[threadIdx.x + d]; sh2[threadIdx.x] += sh2[threadIdx.x + d]; }
    __syncthreads();
  }
  if (threadIdx.x == 0) {
    float m = sh1[0] / (float)tot;
    float var = sh2[0] / (float)tot - m * m;
    meanv[blockIdx.x] = m;
    rstdv[blockIdx.x] = rsqrtf(var + 1e-5f);
  }
}

// ---------------------------------------------------------------------------
// K2: weights f32 -> f16.
// ---------------------------------------------------------------------------
__global__ void k_wcvt(const float* __restrict__ wq, const float* __restrict__ wk,
                       const float* __restrict__ wv, const float* __restrict__ wp,
                       h16* __restrict__ w16) {
  int idx = blockIdx.x * 256 + threadIdx.x;
  int which = idx >> 16;
  int off = idx & 65535;
  const float* src = (which == 0) ? wq : (which == 1) ? wk : (which == 2) ? wv : wp;
  w16[idx] = (h16)src[off];
}

// ---------------------------------------------------------------------------
// K3: normalized activations h f16 [B,N,C] from x [B,C,N] (tiled transpose).
// ---------------------------------------------------------------------------
__global__ void k_hnorm(const float* __restrict__ x,
                        const float* __restrict__ meanv, const float* __restrict__ rstdv,
                        const float* __restrict__ gsc, const float* __restrict__ gbi,
                        h16* __restrict__ h) {
  __shared__ float tile[32][65];
  const int n0 = blockIdx.x * 64;
  const int c0 = blockIdx.y * 32;
  const int b  = blockIdx.z;
  const int t  = threadIdx.x;
  {
    const int nn = t & 63, cb = t >> 6;
#pragma unroll
    for (int i = 0; i < 8; ++i) {
      int cc = cb + i * 4;
      tile[cc][nn] = x[((size_t)(b * CCH + c0 + cc)) * NTOK + n0 + nn];
    }
  }
  __syncthreads();
  {
    const int ci = t & 31, nb = t >> 5;
    const int c = c0 + ci;
    const int g = c >> 3;
    const float m = meanv[b * NGRP + g];
    const float r = rstdv[b * NGRP + g];
    const float sc = gsc[c], bi = gbi[c];
#pragma unroll
    for (int i = 0; i < 8; ++i) {
      int nl = nb + i * 8;
      float v = (tile[ci][nl] - m) * r * sc + bi;
      h[((size_t)b * NTOK + n0 + nl) * CCH + c] = (h16)v;
    }
  }
}

// ---------------------------------------------------------------------------
// K4: q,k GEMM with explicit 2-deep B-fragment pipeline.
// ---------------------------------------------------------------------------
__global__ void __launch_bounds__(256, 1)
k_qkgemm(const h16* __restrict__ h,
         const h16* __restrict__ wq16, const h16* __restrict__ wk16,
         const float* __restrict__ bq, const float* __restrict__ bk,
         h16* __restrict__ qout, h16* __restrict__ kout) {
  const int b = blockIdx.z;
  const int lane = threadIdx.x & 31;
  const int w = threadIdx.x >> 5;
  const int n0 = blockIdx.x * 128 + w * 16;
  const h16* W = (blockIdx.y == 0) ? wq16 : wk16;
  const float* bias = (blockIdx.y == 0) ? bq : bk;
  h16* out = (blockIdx.y == 0) ? qout : kout;
  const h16* hB = h + (size_t)b * NTOK * CCH;

  v8f acc[16];
#pragma unroll
  for (int ot = 0; ot < 16; ++ot) acc[ot] = (v8f)0.f;

  v16h Af[8];
#pragma unroll
  for (int ch = 0; ch < 8; ++ch)
    Af[ch] = load_a_frag(hB + (size_t)n0 * CCH + ch * 32, CCH, lane);

#pragma unroll
  for (int g = 0; g < 4; ++g) {                 // 4 output tiles per group
    const h16* Wg = W + (size_t)(g * 4) * 16 * CCH;
    v16h B0[4], B1[4];
#pragma unroll
    for (int j = 0; j < 4; ++j) B0[j] = load_b_frag(Wg + (size_t)j * 16 * CCH, CCH, lane);
#pragma unroll
    for (int ch = 0; ch < 8; ch += 2) {
#pragma unroll
      for (int j = 0; j < 4; ++j)
        B1[j] = load_b_frag(Wg + (size_t)j * 16 * CCH + (ch + 1) * 32, CCH, lane);
#pragma unroll
      for (int j = 0; j < 4; ++j) acc[g * 4 + j] = wmma_f16(Af[ch], B0[j], acc[g * 4 + j]);
      if (ch + 2 < 8) {
#pragma unroll
        for (int j = 0; j < 4; ++j)
          B0[j] = load_b_frag(Wg + (size_t)j * 16 * CCH + (ch + 2) * 32, CCH, lane);
      }
#pragma unroll
      for (int j = 0; j < 4; ++j) acc[g * 4 + j] = wmma_f16(Af[ch + 1], B1[j], acc[g * 4 + j]);
    }
  }
  const int colb = lane & 15;
  const int rofs = (lane >> 4) << 3;
#pragma unroll
  for (int ot = 0; ot < 16; ++ot) {
    float bb = bias[ot * 16 + colb];
#pragma unroll
    for (int r = 0; r < 8; ++r) {
      int row = n0 + r + rofs;
      out[((size_t)b * NTOK + row) * CCH + ot * 16 + colb] = (h16)(acc[ot][r] + bb);
    }
  }
}

// ---------------------------------------------------------------------------
// K5: vT GEMM: vT[o,n] = sum_c wv[o,c] h[n,c] + bv[o], f16 out [B,C,N].
// ---------------------------------------------------------------------------
__global__ void k_vTgemm(const h16* __restrict__ h, const h16* __restrict__ wv16,
                         const float* __restrict__ bv, h16* __restrict__ vT) {
  const int b = blockIdx.y;
  const int lane = threadIdx.x & 31;
  const int w = threadIdx.x >> 5;
  const int n0 = blockIdx.x * 64;
  const h16* hB = h + (size_t)b * NTOK * CCH;
  h16* vB = vT + (size_t)b * CCH * NTOK;

  v8f acc[2][4];
#pragma unroll
  for (int a = 0; a < 2; ++a)
#pragma unroll
    for (int nt = 0; nt < 4; ++nt) acc[a][nt] = (v8f)0.f;

  v16h Bc[4], Bn[4];
#pragma unroll
  for (int nt = 0; nt < 4; ++nt)
    Bc[nt] = load_b_frag(hB + (size_t)(n0 + nt * 16) * CCH, CCH, lane);
#pragma unroll
  for (int ch = 0; ch < 8; ch += 2) {
    v16h A0 = load_a_frag(wv16 + (size_t)(w * 16) * CCH + ch * 32, CCH, lane);
    v16h A1 = load_a_frag(wv16 + (size_t)((w + 8) * 16) * CCH + ch * 32, CCH, lane);
#pragma unroll
    for (int nt = 0; nt < 4; ++nt)
      Bn[nt] = load_b_frag(hB + (size_t)(n0 + nt * 16) * CCH + (ch + 1) * 32, CCH, lane);
#pragma unroll
    for (int nt = 0; nt < 4; ++nt) {
      acc[0][nt] = wmma_f16(A0, Bc[nt], acc[0][nt]);
      acc[1][nt] = wmma_f16(A1, Bc[nt], acc[1][nt]);
    }
    v16h A2 = load_a_frag(wv16 + (size_t)(w * 16) * CCH + (ch + 1) * 32, CCH, lane);
    v16h A3 = load_a_frag(wv16 + (size_t)((w + 8) * 16) * CCH + (ch + 1) * 32, CCH, lane);
    if (ch + 2 < 8) {
#pragma unroll
      for (int nt = 0; nt < 4; ++nt)
        Bc[nt] = load_b_frag(hB + (size_t)(n0 + nt * 16) * CCH + (ch + 2) * 32, CCH, lane);
    }
#pragma unroll
    for (int nt = 0; nt < 4; ++nt) {
      acc[0][nt] = wmma_f16(A2, Bn[nt], acc[0][nt]);
      acc[1][nt] = wmma_f16(A3, Bn[nt], acc[1][nt]);
    }
  }
  const int colb = lane & 15;
  const int rofs = (lane >> 4) << 3;
#pragma unroll
  for (int a = 0; a < 2; ++a) {
    int o0 = (a == 0 ? w : w + 8) * 16;
#pragma unroll
    for (int r = 0; r < 8; ++r) {
      int o = o0 + r + rofs;
      float bb = bv[o];
#pragma unroll
      for (int nt = 0; nt < 4; ++nt) {
        vB[(size_t)o * NTOK + n0 + nt * 16 + colb] = (h16)(acc[a][nt][r] + bb);
      }
    }
  }
}

// ---------------------------------------------------------------------------
// K6: flash attention with double-buffered TDM staging.
// grid = (N/128, B), 8 waves x 16 query rows. Dynamic LDS:
//   Kt[2]: 32x264 halves (16896B each), Vt[2]: 256x40 halves (20480B each),
//   Pw: 8 x 16x32 halves (8192B). Total 82944B.
// Pipeline: issue TDM for step+1 into the free buffer, s_wait_tensorcnt 0x2
// (per-wave in-order completion => current pair landed), barrier, compute.
// ---------------------------------------------------------------------------
__global__ void __launch_bounds__(256, 1)
k_attn(const h16* __restrict__ q, const h16* __restrict__ k,
       const h16* __restrict__ vT, h16* __restrict__ hatt) {
  extern __shared__ __attribute__((aligned(16))) char smem[];
  h16* const KtB0 = (h16*)(smem);
  h16* const KtB1 = (h16*)(smem + 16896);
  h16* const VtB0 = (h16*)(smem + 33792);
  h16* const VtB1 = (h16*)(smem + 54272);

  const int b = blockIdx.y;
  const int tid = threadIdx.x;
  const int lane = tid & 31;
  const int w = tid >> 5;
  const int qrow0 = blockIdx.x * 128 + w * 16;
  h16* const Pw = (h16*)(smem + 74752) + w * 512;   // 16x32 per wave

  const h16* qg = q  + (size_t)b * NTOK * CCH;
  const h16* kg = k  + (size_t)b * NTOK * CCH;
  const h16* vg = vT + (size_t)b * CCH * NTOK;
  h16* og = hatt + (size_t)b * NTOK * CCH;

#ifdef USE_TDM
  const unsigned ldsK[2] = { lds_addr_of(KtB0), lds_addr_of(KtB1) };
  const unsigned ldsV[2] = { lds_addr_of(VtB0), lds_addr_of(VtB1) };
#endif

  auto stage = [&](int buf, int step) {
    const int key0 = step * 32;
#ifdef USE_TDM
    if (w == 0) {
      // K tile: 32 rows x 256 halves; LDS rows padded 512B -> 528B.
      tdm_load_2d(ldsK[buf], kg + (size_t)key0 * CCH, 256, 32, 256, NTOK, 256, 6, 3);
      // V tile: 256 rows x 32 halves; LDS rows padded 64B -> 80B.
      tdm_load_2d(ldsV[buf], vg + key0, 32, 256, NTOK, 256, NTOK, 3, 3);
    }
#else
    h16* Kt = buf ? KtB1 : KtB0;
    h16* Vt = buf ? VtB1 : VtB0;
#pragma unroll
    for (int i = 0; i < 4; ++i) {
      int u = tid + i * 256;
      int row = u >> 5, c8 = u & 31;
      *(v8h*)(Kt + row * 264 + c8 * 8) = *(const v8h*)(kg + (size_t)(key0 + row) * CCH + c8 * 8);
    }
#pragma unroll
    for (int i = 0; i < 4; ++i) {
      int u = tid + i * 256;
      int row = u >> 2, c8 = u & 3;
      *(v8h*)(Vt + row * 40 + c8 * 8) = *(const v8h*)(vg + (size_t)row * NTOK + key0 + c8 * 8);
    }
#endif
  };

  v16h Qf[8];
#pragma unroll
  for (int ch = 0; ch < 8; ++ch)
    Qf[ch] = load_a_frag(qg + (size_t)qrow0 * CCH + ch * 32, CCH, lane);

  v8f O[16];
#pragma unroll
  for (int ct = 0; ct < 16; ++ct) O[ct] = (v8f)0.f;
  float m_run[8], l_run[8];
#pragma unroll
  for (int r = 0; r < 8; ++r) { m_run[r] = -1e30f; l_run[r] = 0.f; }

  const float sc = 0.0625f * 1.44269504f;  // C^-0.5 * log2(e)
  const int colb = lane & 15;
  const int rofs = (lane >> 4) << 3;

  stage(0, 0);

  for (int step = 0; step < NTOK / 32; ++step) {
    const int cur = step & 1;
    __syncthreads();                       // buffer cur^1 free for overwrite
    if (step + 1 < NTOK / 32) stage(cur ^ 1, step + 1);
#ifdef USE_TDM
    if (w == 0) {
      if (step + 1 < NTOK / 32) asm volatile("s_wait_tensorcnt 0x2" ::: "memory");
      else                      asm volatile("s_wait_tensorcnt 0x0" ::: "memory");
    }
#endif
    __syncthreads();                       // current buffer data visible
    const h16* Kt = cur ? KtB1 : KtB0;
    const h16* Vt = cur ? VtB1 : VtB0;

    // Scores: two 16x16 tiles over this step's 32 keys.
    v8f S0 = (v8f)0.f, S1 = (v8f)0.f;
#pragma unroll
    for (int cp = 0; cp < 4; ++cp) {
      v16h Kb[4];
      Kb[0] = load_b_frag(Kt + (2 * cp) * 32, 264, lane);
      Kb[1] = load_b_frag(Kt + 16 * 264 + (2 * cp) * 32, 264, lane);
      Kb[2] = load_b_frag(Kt + (2 * cp + 1) * 32, 264, lane);
      Kb[3] = load_b_frag(Kt + 16 * 264 + (2 * cp + 1) * 32, 264, lane);
      S0 = wmma_f16(Qf[2 * cp], Kb[0], S0);
      S1 = wmma_f16(Qf[2 * cp], Kb[1], S1);
      S0 = wmma_f16(Qf[2 * cp + 1], Kb[2], S0);
      S1 = wmma_f16(Qf[2 * cp + 1], Kb[3], S1);
    }

    // Online softmax per row (row r lives in element r across a 16-lane half).
    float alpha[8], p0[8], p1[8];
#pragma unroll
    for (int r = 0; r < 8; ++r) {
      float raw = fmaxf(S0[r], S1[r]);
      raw = half_max16(raw);
      float mnew = fmaxf(m_run[r], raw * sc);
      alpha[r] = exp2f(m_run[r] - mnew);
      m_run[r] = mnew;
      p0[r] = exp2f(S0[r] * sc - mnew);
      p1[r] = exp2f(S1[r] * sc - mnew);
      float rs = half_sum16(p0[r] + p1[r]);
      l_run[r] = l_run[r] * alpha[r] + rs;
    }
#pragma unroll
    for (int ct = 0; ct < 16; ++ct)
#pragma unroll
      for (int r = 0; r < 8; ++r) O[ct][r] *= alpha[r];

    // Transpose P (C-layout regs -> A-layout) through per-wave LDS scratch.
#pragma unroll
    for (int r = 0; r < 8; ++r) {
      int row = r + rofs;
      Pw[row * 32 + colb]      = (h16)p0[r];
      Pw[row * 32 + 16 + colb] = (h16)p1[r];
    }
    v16h Pf = load_a_frag(Pw, 32, lane);

    // O += P * V (16 channel tiles, batched LDS frag loads).
#pragma unroll
    for (int g = 0; g < 4; ++g) {
      v16h Vb[4];
#pragma unroll
      for (int j = 0; j < 4; ++j)
        Vb[j] = load_b_frag(Vt + (size_t)(g * 4 + j) * 16 * 40, 40, lane);
#pragma unroll
      for (int j = 0; j < 4; ++j)
        O[g * 4 + j] = wmma_f16(Pf, Vb[j], O[g * 4 + j]);
    }
  }

  float inv[8];
#pragma unroll
  for (int r = 0; r < 8; ++r) inv[r] = 1.f / l_run[r];
#pragma unroll
  for (int ct = 0; ct < 16; ++ct) {
#pragma unroll
    for (int r = 0; r < 8; ++r) {
      int row = qrow0 + r + rofs;
      og[(size_t)row * CCH + ct * 16 + colb] = (h16)(O[ct][r] * inv[r]);
    }
  }
}

// ---------------------------------------------------------------------------
// K7: projection + residual (f32 out), pipelined like K5.
// ---------------------------------------------------------------------------
__global__ void k_proj(const h16* __restrict__ hatt, const h16* __restrict__ wp16,
                       const float* __restrict__ bp, const float* __restrict__ x,
                       float* __restrict__ out) {
  const int b = blockIdx.y;
  const int lane = threadIdx.x & 31;
  const int w = threadIdx.x >> 5;
  const int n0 = blockIdx.x * 64;
  const h16* hB = hatt + (size_t)b * NTOK * CCH;

  v8f acc[2][4];
#pragma unroll
  for (int a = 0; a < 2; ++a)
#pragma unroll
    for (int nt = 0; nt < 4; ++nt) acc[a][nt] = (v8f)0.f;

  v16h Bc[4], Bn[4];
#pragma unroll
  for (int nt = 0; nt < 4; ++nt)
    Bc[nt] = load_b_frag(hB + (size_t)(n0 + nt * 16) * CCH, CCH, lane);
#pragma unroll
  for (int ch = 0; ch < 8; ch += 2) {
    v16h A0 = load_a_frag(wp16 + (size_t)(w * 16) * CCH + ch * 32, CCH, lane);
    v16h A1 = load_a_frag(wp16 + (size_t)((w + 8) * 16) * CCH + ch * 32, CCH, lane);
#pragma unroll
    for (int nt = 0; nt < 4; ++nt)
      Bn[nt] = load_b_frag(hB + (size_t)(n0 + nt * 16) * CCH + (ch + 1) * 32, CCH, lane);
#pragma unroll
    for (int nt = 0; nt < 4; ++nt) {
      acc[0][nt] = wmma_f16(A0, Bc[nt], acc[0][nt]);
      acc[1][nt] = wmma_f16(A1, Bc[nt], acc[1][nt]);
    }
    v16h A2 = load_a_frag(wp16 + (size_t)(w * 16) * CCH + (ch + 1) * 32, CCH, lane);
    v16h A3 = load_a_frag(wp16 + (size_t)((w + 8) * 16) * CCH + (ch + 1) * 32, CCH, lane);
    if (ch + 2 < 8) {
#pragma unroll
      for (int nt = 0; nt < 4; ++nt)
        Bc[nt] = load_b_frag(hB + (size_t)(n0 + nt * 16) * CCH + (ch + 2) * 32, CCH, lane);
    }
#pragma unroll
    for (int nt = 0; nt < 4; ++nt) {
      acc[0][nt] = wmma_f16(A2, Bn[nt], acc[0][nt]);
      acc[1][nt] = wmma_f16(A3, Bn[nt], acc[1][nt]);
    }
  }
  const int colb = lane & 15;
  const int rofs = (lane >> 4) << 3;
#pragma unroll
  for (int a = 0; a < 2; ++a) {
    int o0 = (a == 0 ? w : w + 8) * 16;
#pragma unroll
    for (int r = 0; r < 8; ++r) {
      int o = o0 + r + rofs;
      float bb = bp[o];
#pragma unroll
      for (int nt = 0; nt < 4; ++nt) {
        size_t idx = ((size_t)(b * CCH + o)) * NTOK + n0 + nt * 16 + colb;
        out[idx] = x[idx] + acc[a][nt][r] + bb;
      }
    }
  }
}

// ---------------------------------------------------------------------------
extern "C" void kernel_launch(void* const* d_in, const int* in_sizes, int n_in,
                              void* d_out, int out_size, void* d_ws, size_t ws_size,
                              hipStream_t stream) {
  const float* x   = (const float*)d_in[0];
  const float* gsc = (const float*)d_in[1];
  const float* gbi = (const float*)d_in[2];
  const float* wq  = (const float*)d_in[3];
  const float* bq  = (const float*)d_in[4];
  const float* wk  = (const float*)d_in[5];
  const float* bk  = (const float*)d_in[6];
  const float* wv  = (const float*)d_in[7];
  const float* bv  = (const float*)d_in[8];
  const float* wp  = (const float*)d_in[9];
  const float* bp  = (const float*)d_in[10];
  float* out = (float*)d_out;

  char* p = (char*)d_ws;
  auto alloc = [&](size_t bytes) { void* r = (void*)p; p += (bytes + 255) & ~(size_t)255; return r; };
  float* meanv = (float*)alloc(BATCH * NGRP * sizeof(float));
  float* rstdv = (float*)alloc(BATCH * NGRP * sizeof(float));
  h16* w16   = (h16*)alloc((size_t)4 * CCH * CCH * sizeof(h16));
  h16* hbuf  = (h16*)alloc((size_t)BATCH * NTOK * CCH * sizeof(h16));
  h16* qbuf  = (h16*)alloc((size_t)BATCH * NTOK * CCH * sizeof(h16));
  h16* kbuf  = (h16*)alloc((size_t)BATCH * NTOK * CCH * sizeof(h16));
  h16* vTbuf = (h16*)alloc((size_t)BATCH * CCH * NTOK * sizeof(h16));
  h16* abuf  = (h16*)alloc((size_t)BATCH * NTOK * CCH * sizeof(h16));
  h16* wq16 = w16;
  h16* wk16 = w16 + (size_t)CCH * CCH;
  h16* wv16 = w16 + (size_t)2 * CCH * CCH;
  h16* wp16 = w16 + (size_t)3 * CCH * CCH;

  k_gn_stats<<<BATCH * NGRP, 256, 0, stream>>>(x, meanv, rstdv);
  k_wcvt<<<(4 * CCH * CCH) / 256, 256, 0, stream>>>(wq, wk, wv, wp, w16);
  k_hnorm<<<dim3(NTOK / 64, CCH / 32, BATCH), 256, 0, stream>>>(x, meanv, rstdv, gsc, gbi, hbuf);
  k_qkgemm<<<dim3(NTOK / 128, 2, BATCH), 256, 0, stream>>>(hbuf, wq16, wk16, bq, bk, qbuf, kbuf);
  k_vTgemm<<<dim3(NTOK / 64, BATCH), 256, 0, stream>>>(hbuf, wv16, bv, vTbuf);
  k_attn<<<dim3(NTOK / 128, BATCH), 256, 82944, stream>>>(qbuf, kbuf, vTbuf, abuf);
  k_proj<<<dim3(NTOK / 64, BATCH), 256, 0, stream>>>(abuf, wp16, bp, x, out);
}
